// GlobalModel_7662221656191
// MI455X (gfx1250) — matrix operand order repaired and checked by reading, compiled, not run on hardware
//
#include <hip/hip_runtime.h>
#include <hip/hip_bf16.h>
#include <cstdint>

// ---------------------------------------------------------------------------
// Vector types for WMMA (CDNA5 / gfx1250, wave32)
// ---------------------------------------------------------------------------
typedef __attribute__((ext_vector_type(16))) __bf16 v16bf;
typedef __attribute__((ext_vector_type(8)))  __bf16 v8bf;
typedef __attribute__((ext_vector_type(4)))  __bf16 v4bf;
typedef __attribute__((ext_vector_type(8)))  float  v8f;

// Problem constants (match reference)
#define DL   128
#define DG   128
#define DH   256
#define DP   128
#define KDIM 256            // all three GEMMs have K = 256
#define ROWS 128            // node rows per workgroup tile
#define BSTRIDE 264         // bf16 row stride in LDS (padding kills bank conflicts)
#define FSTRIDE 132         // f32  row stride in LDS (aliases the bf16 buffer)
#define LN_EPS 1e-5f

// ---------------------------------------------------------------------------
// GEMM over one 128xK tile resident in LDS.
//   - wave32; 8 waves per block; wave w owns columns [w*NT*16, (w+1)*NT*16)
//   - B fragments (weights) pre-swizzled in global memory in exact
//     V_WMMA_F32_16X16X32_BF16 B-layout: block(kt,nt) = 512 bf16, lane L owns
//     16 contiguous bf16 at blockBase + L*16:
//        element i of lane L  ==  W[kt*32 + (i>>3)*16 + (L>>4)*8 + (i&7)][nt*16 + (L&15)]
//   - A fragments read from LDS per ISA A-layout:
//        lane L, VGPRs 0..3 : k = ks*32 + (L>>4)*8 + {0..7}
//        lane L, VGPRs 4..7 : k = ks*32 + 16 + (L>>4)*8 + {0..7}
//     -> two 16-byte ds_load_b128 per fragment.
//   - C/D layout: lane L col n = (L&15); VGPR j row = j + 8*(L>>4).
// ---------------------------------------------------------------------------
template <int NT, bool RELU_BF16>
__device__ __forceinline__ void gemm256(const __bf16* __restrict__ sIn,
                                        const __bf16* __restrict__ Wp,
                                        const float* __restrict__ bias,
                                        __bf16* __restrict__ outB,
                                        float* __restrict__ outF)
{
    const int lane  = threadIdx.x & 31;
    const int wv    = threadIdx.x >> 5;     // 0..7
    const int l15   = lane & 15;
    const int lhalf = lane >> 4;            // 0/1
    const int nTilesTotal = NT * 8;         // Nout / 16

    // ---- resident B fragments: NT col-tiles x 8 k-steps (loaded once) ----
    v16bf Bf[NT][8];
#pragma unroll
    for (int nt = 0; nt < NT; ++nt) {
#pragma unroll
        for (int ks = 0; ks < 8; ++ks) {
            const int blk = ks * nTilesTotal + (wv * NT + nt);
            Bf[nt][ks] = *(const v16bf*)(Wp + (size_t)blk * 512 + lane * 16);
        }
    }
    float bn[NT];
#pragma unroll
    for (int nt = 0; nt < NT; ++nt)
        bn[nt] = bias[(wv * NT + nt) * 16 + l15];

#pragma unroll 1
    for (int rt = 0; rt < 8; ++rt) {
        v8f acc[NT];
#pragma unroll
        for (int nt = 0; nt < NT; ++nt)
            acc[nt] = (v8f){0.f, 0.f, 0.f, 0.f, 0.f, 0.f, 0.f, 0.f};

        const __bf16* rp = sIn + (size_t)(rt * 16 + l15) * BSTRIDE;
#pragma unroll
        for (int ks = 0; ks < 8; ++ks) {
            const v8bf lo = *(const v8bf*)(rp + ks * 32 + lhalf * 8);
            const v8bf hi = *(const v8bf*)(rp + ks * 32 + 16 + lhalf * 8);
            v16bf a;
#pragma unroll
            for (int i = 0; i < 8; ++i) { a[i] = lo[i]; a[8 + i] = hi[i]; }
#pragma unroll
            for (int nt = 0; nt < NT; ++nt)
                acc[nt] = __builtin_amdgcn_wmma_f32_16x16x32_bf16(
                    /*neg_a=*/false, a, /*neg_b=*/false, Bf[nt][ks],
                    /*c_mod=*/(short)0, acc[nt],
                    /*reuse_a=*/false, /*reuse_b=*/false);
        }

        // ---- epilogue: bias (+ReLU->bf16  |  f32 pre-LayerNorm) ----
#pragma unroll
        for (int nt = 0; nt < NT; ++nt) {
            const int col = (wv * NT + nt) * 16 + l15;
#pragma unroll
            for (int j = 0; j < 8; ++j) {
                const int orow = rt * 16 + j + 8 * lhalf;
                float v = acc[nt][j] + bn[nt];
                if (RELU_BF16) {
                    v = fmaxf(v, 0.0f);
                    outB[(size_t)orow * BSTRIDE + col] = (__bf16)v;
                } else {
                    outF[(size_t)orow * FSTRIDE + col] = v;
                }
            }
        }
    }
}

// ---------------------------------------------------------------------------
// Weight pre-pack: fp32 row-major [K][Nout] -> bf16 B-fragment-order blocks
// ---------------------------------------------------------------------------
__global__ void pack_weights(const float* __restrict__ W, __bf16* __restrict__ out,
                             int K, int Nout)
{
    const int total  = K * Nout;
    const int ntiles = Nout >> 4;
    for (int idx = blockIdx.x * blockDim.x + threadIdx.x; idx < total;
         idx += gridDim.x * blockDim.x) {
        const int blk    = idx >> 9;        // /512
        const int within = idx & 511;
        const int lane   = within >> 4;
        const int i      = within & 15;
        const int kt = blk / ntiles;
        const int nt = blk % ntiles;
        const int k  = kt * 32 + (i >> 3) * 16 + (lane >> 4) * 8 + (i & 7);
        const int n  = nt * 16 + (lane & 15);
        out[idx] = (__bf16)W[(size_t)k * Nout + n];
    }
}

__global__ void zero_f32(float* __restrict__ p, int n)
{
    const int i = blockIdx.x * blockDim.x + threadIdx.x;
    if (i < n) p[i] = 0.0f;
}

// ---------------------------------------------------------------------------
// Node kernel: concat(x, u[batch]) -> MLP(3 GEMMs) -> LayerNorm -> segment sum
// ---------------------------------------------------------------------------
__global__ __launch_bounds__(256) void node_kernel(
    const float* __restrict__ x, const float* __restrict__ u,
    const long long* __restrict__ batch,
    const __bf16* __restrict__ W1, const float* __restrict__ b1,
    const __bf16* __restrict__ W2, const float* __restrict__ b2,
    const __bf16* __restrict__ W3, const float* __restrict__ b3,
    const float* __restrict__ gamma, const float* __restrict__ beta,
    float* __restrict__ agg, int N)
{
    extern __shared__ char smem[];
    __bf16* bufA   = (__bf16*)smem;                 // [128][264] bf16
    __bf16* bufB   = bufA + ROWS * BSTRIDE;         // [128][264] bf16
    float*  outF   = (float*)bufB;                  // aliases bufB: [128][132] f32
    int*    sBatch = (int*)(bufB + ROWS * BSTRIDE); // [128]

    const int base = blockIdx.x * ROWS;

    // stage segment ids (sentinel -1 for rows past N)
    for (int r = threadIdx.x; r < ROWS; r += 256) {
        const int node = base + r;
        sBatch[r] = (node < N) ? (int)batch[node] : -1;
    }
    __syncthreads();

    // hint the weight stream into cache early
    if (threadIdx.x == 0) {
        __builtin_prefetch(W1, 0, 3);
        __builtin_prefetch(W2, 0, 3);
        __builtin_prefetch(W3, 0, 3);
    }

    // stage concat(x, u[batch]) as bf16: 128 rows x 64 float4
    for (int idx = threadIdx.x; idx < ROWS * 64; idx += 256) {
        const int r  = idx >> 6;
        const int c4 = idx & 63;
        const int node = base + r;
        float4 v = make_float4(0.f, 0.f, 0.f, 0.f);
        if (node < N) {
            if (c4 < 32) {
                v = ((const float4*)x)[(size_t)node * 32 + c4];
            } else {
                const int b = sBatch[r];
                v = ((const float4*)u)[(size_t)b * 32 + (c4 - 32)];
            }
        }
        v4bf h;
        h[0] = (__bf16)v.x; h[1] = (__bf16)v.y; h[2] = (__bf16)v.z; h[3] = (__bf16)v.w;
        *(v4bf*)(bufA + (size_t)r * BSTRIDE + c4 * 4) = h;
    }
    __syncthreads();

    gemm256<2, true >(bufA, W1, b1, bufB, nullptr);   // [128,256] relu
    __syncthreads();
    gemm256<2, true >(bufB, W2, b2, bufA, nullptr);   // [128,256] relu
    __syncthreads();
    gemm256<1, false>(bufA, W3, b3, nullptr, outF);   // [128,128] f32
    __syncthreads();

    // per-row LayerNorm (threads 0..127 = rows)
    if (threadIdx.x < ROWS) {
        float* rowp = outF + (size_t)threadIdx.x * FSTRIDE;
        float mu = 0.f;
        for (int c = 0; c < DP; ++c) mu += rowp[c];
        mu *= (1.0f / DP);
        float var = 0.f;
        for (int c = 0; c < DP; ++c) { const float d = rowp[c] - mu; var += d * d; }
        var *= (1.0f / DP);
        const float rstd = rsqrtf(var + LN_EPS);
        for (int c = 0; c < DP; ++c)
            rowp[c] = (rowp[c] - mu) * rstd * gamma[c] + beta[c];
    }
    __syncthreads();

    // segment pre-reduction (batch is sorted) -> few global atomics per tile
    if (threadIdx.x < DP) {
        const int c = threadIdx.x;
        float s = 0.f;
        for (int r = 0; r < ROWS; ++r) {
            s += outF[(size_t)r * FSTRIDE + c];
            const int b  = sBatch[r];
            const int bn = (r < ROWS - 1) ? sBatch[r + 1] : -2;
            if (b != bn) {
                if (b >= 0) atomicAdd(&agg[(size_t)b * DP + c], s);
                s = 0.f;
            }
        }
    }
}

// ---------------------------------------------------------------------------
// Global kernel: concat(agg, u) -> MLP -> LayerNorm -> + u
// ---------------------------------------------------------------------------
__global__ __launch_bounds__(256) void global_kernel(
    const float* __restrict__ agg, const float* __restrict__ u,
    const __bf16* __restrict__ W1, const float* __restrict__ b1,
    const __bf16* __restrict__ W2, const float* __restrict__ b2,
    const __bf16* __restrict__ W3, const float* __restrict__ b3,
    const float* __restrict__ gamma, const float* __restrict__ beta,
    float* __restrict__ out)
{
    extern __shared__ char smem[];
    __bf16* bufA = (__bf16*)smem;
    __bf16* bufB = bufA + ROWS * BSTRIDE;
    float*  outF = (float*)bufB;

    const int base = blockIdx.x * ROWS;

    for (int idx = threadIdx.x; idx < ROWS * 64; idx += 256) {
        const int r  = idx >> 6;
        const int c4 = idx & 63;
        const int row = base + r;
        float4 v = (c4 < 32) ? ((const float4*)agg)[(size_t)row * 32 + c4]
                             : ((const float4*)u)[(size_t)row * 32 + (c4 - 32)];
        v4bf h;
        h[0] = (__bf16)v.x; h[1] = (__bf16)v.y; h[2] = (__bf16)v.z; h[3] = (__bf16)v.w;
        *(v4bf*)(bufA + (size_t)r * BSTRIDE + c4 * 4) = h;
    }
    __syncthreads();

    gemm256<2, true >(bufA, W1, b1, bufB, nullptr);
    __syncthreads();
    gemm256<2, true >(bufB, W2, b2, bufA, nullptr);
    __syncthreads();
    gemm256<1, false>(bufA, W3, b3, nullptr, outF);
    __syncthreads();

    if (threadIdx.x < ROWS) {
        const int r   = threadIdx.x;
        const int row = base + r;
        float* rowp = outF + (size_t)r * FSTRIDE;
        float mu = 0.f;
        for (int c = 0; c < DG; ++c) mu += rowp[c];
        mu *= (1.0f / DG);
        float var = 0.f;
        for (int c = 0; c < DG; ++c) { const float d = rowp[c] - mu; var += d * d; }
        var *= (1.0f / DG);
        const float rstd = rsqrtf(var + LN_EPS);
        for (int c = 0; c < DG; ++c)
            out[(size_t)row * DG + c] =
                (rowp[c] - mu) * rstd * gamma[c] + beta[c] + u[(size_t)row * DG + c];
    }
}

// ---------------------------------------------------------------------------
// Host launcher
// ---------------------------------------------------------------------------
extern "C" void kernel_launch(void* const* d_in, const int* in_sizes, int n_in,
                              void* d_out, int out_size, void* d_ws, size_t ws_size,
                              hipStream_t stream)
{
    const float*     x     = (const float*)d_in[0];
    const float*     u     = (const float*)d_in[1];
    const long long* batch = (const long long*)d_in[2];
    const float* W1a = (const float*)d_in[3];  const float* b1a = (const float*)d_in[4];
    const float* W2a = (const float*)d_in[5];  const float* b2a = (const float*)d_in[6];
    const float* W3a = (const float*)d_in[7];  const float* b3a = (const float*)d_in[8];
    const float* g_a = (const float*)d_in[9];  const float* be_a = (const float*)d_in[10];
    const float* W1b = (const float*)d_in[11]; const float* b1b = (const float*)d_in[12];
    const float* W2b = (const float*)d_in[13]; const float* b2b = (const float*)d_in[14];
    const float* W3b = (const float*)d_in[15]; const float* b3b = (const float*)d_in[16];
    const float* g_b = (const float*)d_in[17]; const float* be_b = (const float*)d_in[18];

    const int N = in_sizes[0] / DL;     // 500000
    const int B = in_sizes[1] / DG;     // 1024

    // workspace layout
    char*  ws  = (char*)d_ws;
    float* agg = (float*)ws;
    size_t off = ((size_t)B * DP * sizeof(float) + 255) & ~(size_t)255;
    __bf16* pW1a = (__bf16*)(ws + off); off += (size_t)KDIM * DH * 2;
    __bf16* pW2a = (__bf16*)(ws + off); off += (size_t)KDIM * DH * 2;
    __bf16* pW3a = (__bf16*)(ws + off); off += (size_t)KDIM * DP * 2;
    __bf16* pW1b = (__bf16*)(ws + off); off += (size_t)KDIM * DH * 2;
    __bf16* pW2b = (__bf16*)(ws + off); off += (size_t)KDIM * DH * 2;
    __bf16* pW3b = (__bf16*)(ws + off); off += (size_t)KDIM * DG * 2;

    zero_f32<<<(B * DP + 255) / 256, 256, 0, stream>>>(agg, B * DP);
    pack_weights<<<128, 256, 0, stream>>>(W1a, pW1a, KDIM, DH);
    pack_weights<<<128, 256, 0, stream>>>(W2a, pW2a, KDIM, DH);
    pack_weights<<<64,  256, 0, stream>>>(W3a, pW3a, KDIM, DP);
    pack_weights<<<128, 256, 0, stream>>>(W1b, pW1b, KDIM, DH);
    pack_weights<<<128, 256, 0, stream>>>(W2b, pW2b, KDIM, DH);
    pack_weights<<<64,  256, 0, stream>>>(W3b, pW3b, KDIM, DG);

    const size_t smem = (size_t)2 * ROWS * BSTRIDE * sizeof(__bf16)
                      + ROWS * sizeof(int);

    node_kernel<<<(N + ROWS - 1) / ROWS, 256, smem, stream>>>(
        x, u, batch, pW1a, b1a, pW2a, b2a, pW3a, b3a, g_a, be_a, agg, N);

    global_kernel<<<B / ROWS, 256, smem, stream>>>(
        agg, u, pW1b, b1b, pW2b, b2b, pW3b, b3b, g_b, be_b, (float*)d_out);
}
